// PointTransformerBlock_44332652430155
// MI455X (gfx1250) — matrix operand order repaired and checked by reading, not compiled, not run on hardware
//
#include <hip/hip_runtime.h>

// Point Transformer block for MI455X (gfx1250, wave32, WMMA).
//   K0 prep_frags : pre-swizzle 7 weight mats (f32 row-major -> f16 B-fragment order)
//   K1 knn        : brute-force top-16 neighbors (LDS-tiled candidate scan)
//   K2 gemm_proj3 : Q/KX/VX = x@{Wq,Wk,Wv}; A loaded+converted once, B frags in LDS
//   K3 fused_attn : 1 wave/point; weight frags staged in LDS (96 KB, ds_read_b128),
//                   pe MLP, h = q - kx[ind] + pe, mid MLP, softmax-K, agg
//   K4 gemm_final : out = agg@fin_w + fin_b + x; fin_w frags in LDS

#define N_PTS 16384
#define DIM   128
#define KNBR  16
#define WAVES 8
#define TP    136   // LDS tile pitch in halves (272B row, 16B aligned)

typedef __attribute__((ext_vector_type(16))) _Float16 v16h;
typedef __attribute__((ext_vector_type(8)))  float    v8f;

#define MAT_WQ   0
#define MAT_WK   1
#define MAT_WV   2
#define MAT_PEW2 3
#define MAT_MID1 4
#define MAT_MID2 5
#define MAT_FIN  6

#define FRAG_HALVES_PER_MAT (8 * 4 * 32 * 16)   // 16384 halves = 32 KB

__device__ __forceinline__ v8f wmma16(v16h a, v16h b, v8f c) {
  return __builtin_amdgcn_wmma_f32_16x16x32_f16(
      false, a, false, b, (short)0, c, false, false);
}

// A-fragment (16x32 f16, ISA 7.12.2): lane row = l%16,
// halves 0-7 -> K = kc*32 + 8*(l/16) + i ; halves 8-15 -> +16.
__device__ __forceinline__ v16h load_a_frag(const _Float16* tile, int lane, int kc) {
  int row = lane & 15, half = lane >> 4;
  const _Float16* p = tile + row * TP + kc * 32 + 8 * half;
  union { v16h v; int4 q[2]; } u;
  u.q[0] = *(const int4*)(p);
  u.q[1] = *(const int4*)(p + 16);
  return u.v;
}

// B-fragment (32x16 f16): lane col = l%16, halves 0-7 -> K = kc*32 + 8*(l/16) + i ; halves 8-15 -> +16.
// Pre-swizzled so each lane reads 32 contiguous bytes (global or LDS pointer).
__device__ __forceinline__ v16h load_b_frag(const _Float16* base, int mat, int t,
                                            int kc, int lane) {
  const _Float16* p = base + (size_t)(((mat * 8 + t) * 4 + kc) * 32 + lane) * 16;
  union { v16h v; int4 q[2]; } u;
  u.q[0] = ((const int4*)p)[0];
  u.q[1] = ((const int4*)p)[1];
  return u.v;
}

// A-fragment straight from a row-major f32 matrix (lda = 128), cvt to f16.
__device__ __forceinline__ v16h load_a_f32(const float* A, int row, int kc, int half) {
  const float* p = A + (size_t)row * DIM + kc * 32 + 8 * half;
  v16h r;
#pragma unroll
  for (int i = 0; i < 8; ++i) r[i] = (_Float16)p[i];
#pragma unroll
  for (int i = 0; i < 8; ++i) r[8 + i] = (_Float16)p[16 + i];
  return r;
}

// ---------------- K0: weight fragment pre-swizzle --------------------------
__global__ __launch_bounds__(256) void prep_frags(
    const float* W0, const float* W1, const float* W2, const float* W3,
    const float* W4, const float* W5, const float* W6, _Float16* frag) {
  int id = blockIdx.x * 256 + threadIdx.x;   // 0..1023 per matrix
  if (id >= 1024) return;
  int mat = blockIdx.y;
  const float* W;
  switch (mat) {
    case 0: W = W0; break; case 1: W = W1; break; case 2: W = W2; break;
    case 3: W = W3; break; case 4: W = W4; break; default: W = (mat == 5) ? W5 : W6;
  }
  int lane = id & 31, kc = (id >> 5) & 3, t = id >> 7;
  int n = t * 16 + (lane & 15), half = lane >> 4;
  _Float16* dst = frag + (size_t)(((mat * 8 + t) * 4 + kc) * 32 + lane) * 16;
#pragma unroll
  for (int i = 0; i < 16; ++i)
    dst[i] = (_Float16)W[(size_t)(kc * 32 + 8 * half + (i & 7) + 16 * (i >> 3)) * DIM + n];
}

// ---------------- K1: brute-force kNN --------------------------------------
__global__ __launch_bounds__(256) void knn_kernel(const float* __restrict__ pos,
                                                  int* __restrict__ ind) {
  __shared__ float sp[256 * 3];
  int i = blockIdx.x * 256 + threadIdx.x;
  float px = pos[i * 3 + 0], py = pos[i * 3 + 1], pz = pos[i * 3 + 2];
  float bd[KNBR]; int bi[KNBR];
#pragma unroll
  for (int k = 0; k < KNBR; ++k) { bd[k] = 3.0e38f; bi[k] = 0; }
  for (int tile = 0; tile < N_PTS / 256; ++tile) {
    __syncthreads();
    int c = tile * 256 + threadIdx.x;
    sp[threadIdx.x * 3 + 0] = pos[c * 3 + 0];
    sp[threadIdx.x * 3 + 1] = pos[c * 3 + 1];
    sp[threadIdx.x * 3 + 2] = pos[c * 3 + 2];
    __syncthreads();
    for (int j = 0; j < 256; ++j) {
      float dx = px - sp[j * 3 + 0], dy = py - sp[j * 3 + 1], dz = pz - sp[j * 3 + 2];
      float d2 = dx * dx + dy * dy + dz * dz;
      if (d2 < bd[KNBR - 1]) {
        bd[KNBR - 1] = d2; bi[KNBR - 1] = tile * 256 + j;
#pragma unroll
        for (int s = KNBR - 1; s > 0; --s)
          if (bd[s] < bd[s - 1]) {
            float tf = bd[s]; bd[s] = bd[s - 1]; bd[s - 1] = tf;
            int ti = bi[s];   bi[s] = bi[s - 1]; bi[s - 1] = ti;
          }
      }
    }
  }
#pragma unroll
  for (int k = 0; k < KNBR; ++k) ind[i * KNBR + k] = bi[k];
}

// ---------------- K2: fused Q/K/V projection (B frags staged in LDS) -------
#define PROJ_SMEM_BYTES (3 * FRAG_HALVES_PER_MAT * 2)   // 96 KB

__global__ __launch_bounds__(256) void gemm_proj3(const float* __restrict__ A,
                                                  const _Float16* __restrict__ frag,
                                                  float* __restrict__ Q,
                                                  float* __restrict__ KX,
                                                  float* __restrict__ VX) {
  extern __shared__ char smemp[];
  _Float16* fragS = (_Float16*)smemp;
  {
    const int4* src = (const int4*)frag;    // mats 0..2 are at offset 0
    int4* dst = (int4*)fragS;
    for (int i = threadIdx.x; i < 3 * FRAG_HALVES_PER_MAT / 8; i += 256)
      dst[i] = src[i];
  }
  int wid = (blockIdx.x * blockDim.x + threadIdx.x) >> 5;
  int lane = threadIdx.x & 31;
  int m0 = wid * 16;
  int col = lane & 15, half = lane >> 4;

  v16h a[4];
#pragma unroll
  for (int kc = 0; kc < 4; ++kc) a[kc] = load_a_f32(A, m0 + col, kc, half);
  __syncthreads();

  float* const outs[3] = {Q, KX, VX};
#pragma unroll
  for (int m = 0; m < 3; ++m) {
    float* C = outs[m];
#pragma unroll
    for (int t = 0; t < 8; ++t) {
      v8f acc = {};
#pragma unroll
      for (int kc = 0; kc < 4; ++kc)
        acc = wmma16(a[kc], load_b_frag(fragS, m, t, kc, lane), acc);
#pragma unroll
      for (int j = 0; j < 8; ++j)
        C[(size_t)(m0 + j + 8 * half) * DIM + t * 16 + col] = acc[j];
    }
  }
}

// ---------------- K4: final projection, bias + residual --------------------
__global__ __launch_bounds__(256) void gemm_final(const float* __restrict__ A,
                                                  const _Float16* __restrict__ frag,
                                                  const float* __restrict__ bias,
                                                  const float* __restrict__ resid,
                                                  float* __restrict__ C) {
  __shared__ _Float16 fragS[FRAG_HALVES_PER_MAT];   // 32 KB
  {
    const int4* src = (const int4*)(frag + (size_t)MAT_FIN * FRAG_HALVES_PER_MAT);
    int4* dst = (int4*)fragS;
    for (int i = threadIdx.x; i < FRAG_HALVES_PER_MAT / 8; i += 256)
      dst[i] = src[i];
  }
  int wid = (blockIdx.x * blockDim.x + threadIdx.x) >> 5;
  int lane = threadIdx.x & 31;
  int m0 = wid * 16;
  int col = lane & 15, half = lane >> 4;

  v16h a[4];
#pragma unroll
  for (int kc = 0; kc < 4; ++kc) a[kc] = load_a_f32(A, m0 + col, kc, half);
  __syncthreads();

#pragma unroll
  for (int t = 0; t < 8; ++t) {
    v8f acc = {};
#pragma unroll
    for (int kc = 0; kc < 4; ++kc)
      acc = wmma16(a[kc], load_b_frag(fragS, 0, t, kc, lane), acc);
    float bv = bias[t * 16 + col];
#pragma unroll
    for (int j = 0; j < 8; ++j) {
      int r = m0 + j + 8 * half;
      C[(size_t)r * DIM + t * 16 + col] =
          acc[j] + bv + resid[(size_t)r * DIM + t * 16 + col];
    }
  }
}

// ---------------- K3: fused per-point attention ----------------------------
// Dynamic LDS layout:
//   fragS : 3 * 16384 halves  (96 KB)  -- pe_w2 / mid_w1 / mid_w2 B-fragments
//   tileS : WAVES*16*TP halves (34 KB) -- per-wave activation tiles
//   relS  : WAVES*16*4 floats          -- relative positions
//   sw1   : 3*DIM floats, sb1 : DIM floats
#define FUSED_SMEM_BYTES                                                      \
  (3 * FRAG_HALVES_PER_MAT * 2 + WAVES * 16 * TP * 2 +                        \
   WAVES * 16 * 4 * 4 + 3 * DIM * 4 + DIM * 4)

__global__ __launch_bounds__(256) void fused_attn(
    const float* __restrict__ pos, const int* __restrict__ ind,
    const float* __restrict__ Q, const float* __restrict__ KX,
    const float* __restrict__ VX, const _Float16* __restrict__ frag,
    const float* __restrict__ pe_w1, const float* __restrict__ pe_b1,
    const float* __restrict__ pe_b2, const float* __restrict__ mid_b1,
    const float* __restrict__ mid_b2, float* __restrict__ AGG) {
  extern __shared__ char smem[];
  _Float16* fragS = (_Float16*)smem;
  _Float16* tileS = fragS + 3 * FRAG_HALVES_PER_MAT;
  float* relS = (float*)(tileS + WAVES * 16 * TP);
  float* sw1 = relS + WAVES * 16 * 4;
  float* sb1 = sw1 + 3 * DIM;

  int lane = threadIdx.x & 31, wv = threadIdx.x >> 5;
  int n = blockIdx.x * WAVES + wv;            // grid sized exactly
  int col = lane & 15, half = lane >> 4;

  // cooperative stage: 96 KB of weight fragments -> LDS (int4 copies)
  {
    const int4* src = (const int4*)(frag + (size_t)MAT_PEW2 * FRAG_HALVES_PER_MAT);
    int4* dst = (int4*)fragS;
    for (int i = threadIdx.x; i < 3 * FRAG_HALVES_PER_MAT / 8; i += 256)
      dst[i] = src[i];
  }
  for (int i = threadIdx.x; i < 3 * DIM; i += 256) sw1[i] = pe_w1[i];
  for (int i = threadIdx.x; i < DIM; i += 256) sb1[i] = pe_b1[i];

  // neighbor row indices for this lane's half (rows half*8 .. half*8+7)
  int rowIdx[8];
#pragma unroll
  for (int j = 0; j < 8; ++j) rowIdx[j] = ind[n * KNBR + j + 8 * half];

  // relative positions (one neighbor per lane, lanes 0..15)
  if (lane < 16) {
    int idx = ind[n * KNBR + lane];
#pragma unroll
    for (int d = 0; d < 3; ++d)
      relS[(wv * 16 + lane) * 4 + d] = pos[n * 3 + d] - pos[idx * 3 + d];
  }
  __syncthreads();

  _Float16* myTile = tileS + wv * 16 * TP;

  // pe layer 1: t = relu(rel @ pe_w1 + pe_b1) -> f16 tile [16][128] in LDS
#pragma unroll
  for (int e = 0; e < 64; ++e) {
    int id = e * 32 + lane;       // 0..2047
    int r = id >> 7, c = id & 127;
    float acc = sb1[c] + relS[(wv * 16 + r) * 4 + 0] * sw1[c]
                       + relS[(wv * 16 + r) * 4 + 1] * sw1[DIM + c]
                       + relS[(wv * 16 + r) * 4 + 2] * sw1[2 * DIM + c];
    myTile[r * TP + c] = (_Float16)fmaxf(acc, 0.0f);
  }

  // per-column constants (C-layout: col = l%16 + 16*t)
  float qv[8], pb2[8], mb1v[8], mb2v[8];
#pragma unroll
  for (int t = 0; t < 8; ++t) {
    qv[t]   = Q[(size_t)n * DIM + t * 16 + col];
    pb2[t]  = pe_b2[t * 16 + col];
    mb1v[t] = mid_b1[t * 16 + col];
    mb2v[t] = mid_b2[t * 16 + col];
  }

  v16h a[4];

  // pe = t @ pe_w2 + pe_b2  (kept in regs for both h and z)
#pragma unroll
  for (int kc = 0; kc < 4; ++kc) a[kc] = load_a_frag(myTile, lane, kc);
  float pe[8][8];
#pragma unroll
  for (int t = 0; t < 8; ++t) {
    v8f acc = {};
#pragma unroll
    for (int kc = 0; kc < 4; ++kc)
      acc = wmma16(a[kc], load_b_frag(fragS, 0, t, kc, lane), acc);
#pragma unroll
    for (int j = 0; j < 8; ++j) pe[t][j] = acc[j] + pb2[t];
  }

  // h = q - kx[ind] + pe  -> f16 tile (same-wave LDS ops are in-order)
#pragma unroll
  for (int t = 0; t < 8; ++t)
#pragma unroll
    for (int j = 0; j < 8; ++j) {
      float kxv = KX[(size_t)rowIdx[j] * DIM + t * 16 + col];
      myTile[(j + 8 * half) * TP + t * 16 + col] = (_Float16)(qv[t] - kxv + pe[t][j]);
    }

  // y1 = relu(h @ mid_w1 + b1) -> f16 tile
#pragma unroll
  for (int kc = 0; kc < 4; ++kc) a[kc] = load_a_frag(myTile, lane, kc);
#pragma unroll
  for (int t = 0; t < 8; ++t) {
    v8f acc = {};
#pragma unroll
    for (int kc = 0; kc < 4; ++kc)
      acc = wmma16(a[kc], load_b_frag(fragS, 1, t, kc, lane), acc);
#pragma unroll
    for (int j = 0; j < 8; ++j)
      myTile[(j + 8 * half) * TP + t * 16 + col] =
          (_Float16)fmaxf(acc[j] + mb1v[t], 0.0f);
  }

  // y2 = y1 @ mid_w2 + b2, then softmax over K (rows) + aggregation, per tile
#pragma unroll
  for (int kc = 0; kc < 4; ++kc) a[kc] = load_a_frag(myTile, lane, kc);
  const float invScale = 0.088388347648318447f;  // 1/sqrt(128)
#pragma unroll
  for (int t = 0; t < 8; ++t) {
    v8f acc = {};
#pragma unroll
    for (int kc = 0; kc < 4; ++kc)
      acc = wmma16(a[kc], load_b_frag(fragS, 2, t, kc, lane), acc);
    float y[8];
#pragma unroll
    for (int j = 0; j < 8; ++j) y[j] = acc[j] + mb2v[t];

    // column-wise max over 16 rows: 8 local VGPRs + cross-half exchange
    float mx = y[0];
#pragma unroll
    for (int j = 1; j < 8; ++j) mx = fmaxf(mx, y[j]);
    mx = fmaxf(mx, __shfl_xor(mx, 16));

    float den = 0.0f, num = 0.0f;
#pragma unroll
    for (int j = 0; j < 8; ++j) {
      float e = __expf((y[j] - mx) * invScale);
      float z = VX[(size_t)rowIdx[j] * DIM + t * 16 + col] + pe[t][j];
      den += e; num += e * z;
    }
    den += __shfl_xor(den, 16);
    num += __shfl_xor(num, 16);
    if (half == 0) AGG[(size_t)n * DIM + t * 16 + col] = num / den;
  }
}

// ---------------------------------------------------------------------------
extern "C" void kernel_launch(void* const* d_in, const int* in_sizes, int n_in,
                              void* d_out, int out_size, void* d_ws, size_t ws_size,
                              hipStream_t stream) {
  const float* x      = (const float*)d_in[0];
  const float* pos    = (const float*)d_in[1];
  const float* Wq     = (const float*)d_in[2];
  const float* Wk     = (const float*)d_in[3];
  const float* Wv     = (const float*)d_in[4];
  const float* mid_w1 = (const float*)d_in[5];
  const float* mid_b1 = (const float*)d_in[6];
  const float* mid_w2 = (const float*)d_in[7];
  const float* mid_b2 = (const float*)d_in[8];
  const float* pe_w1  = (const float*)d_in[9];
  const float* pe_b1  = (const float*)d_in[10];
  const float* pe_w2  = (const float*)d_in[11];
  const float* pe_b2  = (const float*)d_in[12];
  const float* fin_w  = (const float*)d_in[13];
  const float* fin_b  = (const float*)d_in[14];
  float* out = (float*)d_out;

  // workspace layout (~33.3 MB)
  char* w = (char*)d_ws;
  int* ind = (int*)w;            w += (size_t)N_PTS * KNBR * sizeof(int);   // 1 MB
  float* Q   = (float*)w;        w += (size_t)N_PTS * DIM * sizeof(float);  // 8 MB
  float* KX  = (float*)w;        w += (size_t)N_PTS * DIM * sizeof(float);
  float* VX  = (float*)w;        w += (size_t)N_PTS * DIM * sizeof(float);
  float* AGG = (float*)w;        w += (size_t)N_PTS * DIM * sizeof(float);
  _Float16* frag = (_Float16*)w; // 7 * 32 KB of pre-swizzled f16 fragments

  prep_frags<<<dim3(4, 7, 1), 256, 0, stream>>>(Wq, Wk, Wv, pe_w2, mid_w1,
                                                mid_w2, fin_w, frag);
  knn_kernel<<<N_PTS / 256, 256, 0, stream>>>(pos, ind);

  // 1024 row-tiles of 16 rows -> 128 blocks of 8 waves
  gemm_proj3<<<128, 256, PROJ_SMEM_BYTES, stream>>>(x, frag, Q, KX, VX);

  fused_attn<<<N_PTS / WAVES, 256, FUSED_SMEM_BYTES, stream>>>(
      pos, ind, Q, KX, VX, frag, pe_w1, pe_b1, pe_b2, mid_b1, mid_b2, AGG);

  gemm_final<<<128, 256, 0, stream>>>(AGG, frag, fin_b, x, out);
}
